// SequenceAttention_16389595202093
// MI455X (gfx1250) — matrix operand order repaired and checked by reading, not compiled, not run on hardware
//
#include <hip/hip_runtime.h>
#include <hip/hip_bf16.h>

#define IFZ   256
#define SFZ   1280
#define AHZ   8
#define AFZ   64
#define HDM   512      // AHZ*AFZ
#define SEQ   512
#define NRES  20
#define INV_SCALE 0.125f                      // 1/sqrt(64)
#define INV_RESID 0.70710678118654752440f     // 1/sqrt(2)

typedef __attribute__((ext_vector_type(16))) __bf16 v16bf;
typedef __attribute__((ext_vector_type(8)))  float  v8f;

struct U8 { unsigned u[8]; };

__device__ __forceinline__ unsigned short f2bf(float f) {
  unsigned u = __float_as_uint(f);
  unsigned r = 0x7FFFu + ((u >> 16) & 1u);     // round-to-nearest-even
  return (unsigned short)((u + r) >> 16);
}

// A-matrix 16x32 bf16 fragment (ISA 7.12.2). p = row base + k0 + half*8.
// VGPR j holds K pair at ((j&4)?16:0) + (j&3)*2 -> two contiguous 16B groups.
__device__ __forceinline__ v16bf loadA(const unsigned short* p) {
  U8 f;
#pragma unroll
  for (int j = 0; j < 8; ++j)
    f.u[j] = *(const unsigned*)(p + ((j & 4) ? 16 : 0) + ((j & 3) << 1));
  return __builtin_bit_cast(v16bf, f);
}

// B-matrix 32x16 bf16 fragment. p = row base + k0 + half*8; same K order as A:
// VGPR j holds K pair at ((j&4)?16:0) + (j&3)*2 -> two contiguous 16B groups.
__device__ __forceinline__ v16bf loadB(const unsigned short* p) {
  U8 f;
#pragma unroll
  for (int j = 0; j < 8; ++j)
    f.u[j] = *(const unsigned*)(p + ((j & 4) ? 16 : 0) + ((j & 3) << 1));
  return __builtin_bit_cast(v16bf, f);
}

__device__ __forceinline__ float wsum(float v) {
#pragma unroll
  for (int o = 16; o > 0; o >>= 1) v += __shfl_xor(v, o, 32);
  return v;
}
__device__ __forceinline__ float wmaxr(float v) {
#pragma unroll
  for (int o = 16; o > 0; o >>= 1) v = fmaxf(v, __shfl_xor(v, o, 32));
  return v;
}

// ---------------- elementwise / utility kernels ----------------

__global__ void zero_f32(float* p, long n) {
  long i = (long)blockIdx.x * blockDim.x + threadIdx.x;
  if (i < n) p[i] = 0.f;
}

__global__ void cvt_bf16(const float* __restrict__ s, unsigned short* __restrict__ d, long n) {
  long i = (long)blockIdx.x * blockDim.x + threadIdx.x;
  if (i < n) d[i] = f2bf(s[i]);
}

__global__ void gather_cvt_x(const float* __restrict__ x, const int* __restrict__ pm,
                             unsigned short* __restrict__ d, int M) {
  long i = (long)blockIdx.x * blockDim.x + threadIdx.x;
  if (i < (long)M * IFZ) {
    int m = (int)(i / IFZ), j = (int)(i % IFZ);
    d[i] = f2bf(x[(size_t)pm[m] * IFZ + j]);
  }
}

__global__ void make_bidx(const int* __restrict__ batch, const int* __restrict__ pm,
                          int* __restrict__ bidx, int M) {
  int m = blockIdx.x * blockDim.x + threadIdx.x;
  if (m < M) bidx[m] = batch[pm[m]];
}

// V (BS, 512) f32 -> Vt (B, H, D, S) bf16  (B operand for attn*V needs K=s contiguous)
__global__ void v_transpose(const float* __restrict__ v, unsigned short* __restrict__ vt,
                            int B) {
  long i = (long)blockIdx.x * blockDim.x + threadIdx.x;
  long total = (long)B * SEQ * HDM;
  if (i < total) {
    long row = i / HDM; int col = (int)(i % HDM);
    int b = (int)(row / SEQ), s = (int)(row % SEQ);
    int h = col / AFZ, d = col % AFZ;
    vt[(((size_t)(b * AHZ + h) * AFZ) + d) * SEQ + s] = f2bf(v[i]);
  }
}

__global__ void resid_add(float* __restrict__ h, const float* __restrict__ y, long n) {
  long i = (long)blockIdx.x * blockDim.x + threadIdx.x;
  if (i < n) h[i] = (h[i] + y[i]) * INV_RESID;
}

__global__ void scatter_rows(const float* __restrict__ src, const int* __restrict__ pm,
                             float* __restrict__ dst, int M, int L) {
  long i = (long)blockIdx.x * blockDim.x + threadIdx.x;
  if (i < (long)M * L) {
    int m = (int)(i / L), j = (int)(i % L);
    dst[(size_t)pm[m] * L + j] = src[i];
  }
}

// ------------- generic bf16 WMMA GEMM: C = A(MxK) * B(NxK)^T -------------
// One wave per 32x16 output block (2 M-tiles share one B fragment).
// Out-of-range rows are CLAMPED, not zeroed: A-row m only feeds output row m
// and B-row n only feeds output column n, and those stores are masked, so
// clamped loads never corrupt live results. This keeps the K-loop branch-free.
__global__ void gemm_bf16(const unsigned short* __restrict__ A, int lda,
                          const unsigned short* __restrict__ Bm, int ldb,
                          float* __restrict__ C, int ldc,
                          int Md, int Nd, int Kd,
                          const float* __restrict__ bias, int relu, float scale) {
  int lane = threadIdx.x & 31, l15 = lane & 15, half = lane >> 4;
  int m0 = blockIdx.x * 32, n0 = blockIdx.y * 16;
  int ra0 = min(m0 + l15, Md - 1);
  int ra1 = min(m0 + 16 + l15, Md - 1);
  int rb  = min(n0 + l15, Nd - 1);
  const unsigned short* pa0 = A + (size_t)ra0 * lda + half * 8;
  const unsigned short* pa1 = A + (size_t)ra1 * lda + half * 8;
  const unsigned short* pb  = Bm + (size_t)rb * ldb + half * 8;
  v8f acc0 = {}, acc1 = {};
#pragma unroll 4
  for (int k0 = 0; k0 < Kd; k0 += 32) {
    __builtin_prefetch(pb + 128, 0, 1);    // speculative; dropped if past end
    v16bf b  = loadB(pb);
    v16bf a0 = loadA(pa0);
    v16bf a1 = loadA(pa1);
    acc0 = __builtin_amdgcn_wmma_f32_16x16x32_bf16(false, a0, false, b,
                                                   (short)0, acc0, false, false);
    acc1 = __builtin_amdgcn_wmma_f32_16x16x32_bf16(false, a1, false, b,
                                                   (short)0, acc1, false, false);
    pa0 += 32; pa1 += 32; pb += 32;
  }
  int n = n0 + l15;
  if (n < Nd) {
    float bb = bias ? bias[n] : 0.f;
#pragma unroll
    for (int r = 0; r < 8; ++r) {
      int m = m0 + r + 8 * half;
      if (m < Md) {
        float v = acc0[r] * scale + bb;
        if (relu) v = fmaxf(v, 0.f);
        C[(size_t)m * ldc + n] = v;
      }
      int m2 = m + 16;
      if (m2 < Md) {
        float v = acc1[r] * scale + bb;
        if (relu) v = fmaxf(v, 0.f);
        C[(size_t)m2 * ldc + n] = v;
      }
    }
  }
}

// scores[m,s,h] = q[m,h,:].k[b,s,h,:]/8 computed per (h,b), stored only where
// bidx[m]==b, scattered straight into the unpacked_scores output at row pm[m].
__global__ void attn_scores(const unsigned short* __restrict__ Qb,
                            const unsigned short* __restrict__ Kb,
                            const int* __restrict__ bidx, const int* __restrict__ pm,
                            float* __restrict__ out_sc, int Md, int B) {
  int lane = threadIdx.x & 31, l15 = lane & 15, half = lane >> 4;
  int m0 = blockIdx.x * 16, s0 = blockIdx.y * 16;
  int z = blockIdx.z, h = z / B, b = z - h * B;
  const unsigned short* pa = Qb + (size_t)min(m0 + l15, Md - 1) * HDM
                             + h * AFZ + half * 8;
  const unsigned short* pb = Kb + (size_t)(b * SEQ + s0 + l15) * HDM
                             + h * AFZ + half * 8;
  v8f acc = {};
#pragma unroll
  for (int k0 = 0; k0 < AFZ; k0 += 32) {
    v16bf a  = loadA(pa);
    v16bf bf = loadB(pb);
    acc = __builtin_amdgcn_wmma_f32_16x16x32_bf16(false, a, false, bf,
                                                  (short)0, acc, false, false);
    pa += 32; pb += 32;
  }
#pragma unroll
  for (int r = 0; r < 8; ++r) {
    int m = m0 + r + 8 * half, s = s0 + l15;
    if (m < Md && bidx[m] == b)
      out_sc[((size_t)pm[m] * SEQ + s) * AHZ + h] = acc[r] * INV_SCALE;
  }
}

// softmax over s (unmasked, max-subtracted), *mask, renormalize with EPS;
// write bf16 attention weights laid out (H, M, S) for the next WMMA.
__global__ void softmax_w(const float* __restrict__ out_sc, const int* __restrict__ pm,
                          const int* __restrict__ bidx, const float* __restrict__ mask,
                          unsigned short* __restrict__ Wbf, int Md) {
  int idx = blockIdx.x, m = idx >> 3, h = idx & 7;
  int lane = threadIdx.x;
  size_t base = (size_t)pm[m] * SEQ * AHZ + h;
  int b = bidx[m];
  float mx = -3.0e38f;
  for (int s = lane; s < SEQ; s += 32) mx = fmaxf(mx, out_sc[base + (size_t)s * AHZ]);
  mx = wmaxr(mx);
  float sum = 0.f;
  for (int s = lane; s < SEQ; s += 32) sum += expf(out_sc[base + (size_t)s * AHZ] - mx);
  sum = wsum(sum);
  float msum = 0.f;
  for (int s = lane; s < SEQ; s += 32)
    msum += expf(out_sc[base + (size_t)s * AHZ] - mx) / sum * mask[(size_t)b * SEQ + s];
  msum = wsum(msum);
  float inv = 1.f / (msum + 1e-6f);
  for (int s = lane; s < SEQ; s += 32) {
    float w = expf(out_sc[base + (size_t)s * AHZ] - mx) / sum
              * mask[(size_t)b * SEQ + s] * inv;
    Wbf[((size_t)h * Md + m) * SEQ + s] = f2bf(w);
  }
}

// feat[m,h,d] = sum_s w[m,s,h] * v[b,s,h,d]; per (h,b), predicated store.
__global__ void attn_feat(const unsigned short* __restrict__ Wbf,
                          const unsigned short* __restrict__ Vt,
                          const int* __restrict__ bidx,
                          float* __restrict__ feat, int Md, int B) {
  int lane = threadIdx.x & 31, l15 = lane & 15, half = lane >> 4;
  int m0 = blockIdx.x * 16, d0 = blockIdx.y * 16;
  int z = blockIdx.z, h = z / B, b = z - h * B;
  const unsigned short* pa = Wbf + (size_t)h * Md * SEQ
                             + (size_t)min(m0 + l15, Md - 1) * SEQ + half * 8;
  const unsigned short* pb = Vt + (size_t)(b * AHZ + h) * AFZ * SEQ
                             + (size_t)(d0 + l15) * SEQ + half * 8;
  v8f acc = {};
#pragma unroll 4
  for (int k0 = 0; k0 < SEQ; k0 += 32) {
    __builtin_prefetch(pb + 128, 0, 1);
    v16bf a  = loadA(pa);
    v16bf bf = loadB(pb);
    acc = __builtin_amdgcn_wmma_f32_16x16x32_bf16(false, a, false, bf,
                                                  (short)0, acc, false, false);
    pa += 32; pb += 32;
  }
#pragma unroll
  for (int r = 0; r < 8; ++r) {
    int m = m0 + r + 8 * half, d = d0 + l15;
    if (m < Md && bidx[m] == b)
      feat[(size_t)m * HDM + h * AFZ + d] = acc[r];
  }
}

// LayerNorm (population var, eps 1e-5) -> bf16; one wave per row.
__global__ void ln_to_bf16(const float* __restrict__ X, int L,
                           const float* __restrict__ s, const float* __restrict__ b,
                           unsigned short* __restrict__ out) {
  int row = blockIdx.x, lane = threadIdx.x;
  const float* xr = X + (size_t)row * L;
  float sum = 0.f, sq = 0.f;
  for (int j = lane; j < L; j += 32) { float v = xr[j]; sum += v; sq += v * v; }
  sum = wsum(sum); sq = wsum(sq);
  float mean = sum / L, var = sq / L - mean * mean, inv = rsqrtf(var + 1e-5f);
  for (int j = lane; j < L; j += 32)
    out[(size_t)row * L + j] = f2bf((xr[j] - mean) * inv * s[j] + b[j]);
}

// out_x = LN(x + nf/sqrt(2)) * en_s + en_b, all N rows, f32 out.
__global__ void final_ln(const float* __restrict__ x, const float* __restrict__ nf,
                         const float* __restrict__ es, const float* __restrict__ eb,
                         float* __restrict__ outx) {
  int row = blockIdx.x, lane = threadIdx.x;
  const float* xr = x + (size_t)row * IFZ;
  const float* nr = nf + (size_t)row * IFZ;
  float sum = 0.f, sq = 0.f;
  for (int j = lane; j < IFZ; j += 32) {
    float t = xr[j] + nr[j] * INV_RESID; sum += t; sq += t * t;
  }
  sum = wsum(sum); sq = wsum(sq);
  float mean = sum / IFZ, var = sq / IFZ - mean * mean, inv = rsqrtf(var + 1e-5f);
  for (int j = lane; j < IFZ; j += 32) {
    float t = xr[j] + nr[j] * INV_RESID;
    outx[(size_t)row * IFZ + j] = (t - mean) * inv * es[j] + eb[j];
  }
}

// ---------------------------------------------------------------------------

extern "C" void kernel_launch(void* const* d_in, const int* in_sizes, int n_in,
                              void* d_out, int out_size, void* d_ws, size_t ws_size,
                              hipStream_t stream) {
  const float* x    = (const float*)d_in[0];
  const float* emb  = (const float*)d_in[1];
  const float* mask = (const float*)d_in[2];
  const int*   pm   = (const int*)d_in[3];
  const int*   bat  = (const int*)d_in[4];
  const float* Wq   = (const float*)d_in[5];
  const float* Wk   = (const float*)d_in[6];
  const float* Wv   = (const float*)d_in[7];
  const float* agS  = (const float*)d_in[8];
  const float* agB  = (const float*)d_in[9];
  const float* Wag  = (const float*)d_in[10];
  const float* rls  = (const float*)d_in[11];
  const float* rlb  = (const float*)d_in[12];
  const float* rw1  = (const float*)d_in[13];
  const float* rb1  = (const float*)d_in[14];
  const float* rw2  = (const float*)d_in[15];
  const float* rb2  = (const float*)d_in[16];
  const float* hw   = (const float*)d_in[17];
  const float* hb   = (const float*)d_in[18];
  const float* enS  = (const float*)d_in[19];
  const float* enB  = (const float*)d_in[20];

  const int N  = in_sizes[0] / IFZ;
  const int M  = in_sizes[3];
  const int B  = in_sizes[2] / SEQ;
  const int BS = B * SEQ;

  char* ws = (char*)d_ws;
  size_t off = 0;
  auto alloc = [&](size_t bytes) -> void* {
    void* p = ws + off; off = (off + bytes + 255) & ~(size_t)255; return p;
  };

  int*            bidx   = (int*)           alloc((size_t)M * 4);
  unsigned short* Wqb    = (unsigned short*)alloc((size_t)HDM * IFZ * 2);
  unsigned short* Wkb    = (unsigned short*)alloc((size_t)HDM * SFZ * 2);
  unsigned short* Wvb    = (unsigned short*)alloc((size_t)HDM * SFZ * 2);
  unsigned short* Wagb   = (unsigned short*)alloc((size_t)IFZ * HDM * 2);
  unsigned short* w1b    = (unsigned short*)alloc((size_t)3 * IFZ * IFZ * 2);
  unsigned short* w2b    = (unsigned short*)alloc((size_t)3 * IFZ * IFZ * 2);
  unsigned short* hwb    = (unsigned short*)alloc((size_t)NRES * IFZ * 2);
  unsigned short* embb   = (unsigned short*)alloc((size_t)BS * SFZ * 2);
  unsigned short* xqb    = (unsigned short*)alloc((size_t)M * IFZ * 2);
  float*          qf     = (float*)         alloc((size_t)M * HDM * 4);
  unsigned short* qb     = (unsigned short*)alloc((size_t)M * HDM * 2);
  float*          kf     = (float*)         alloc((size_t)BS * HDM * 4);
  unsigned short* kb     = (unsigned short*)alloc((size_t)BS * HDM * 2);
  float*          vf     = (float*)         alloc((size_t)BS * HDM * 4);
  unsigned short* vtb    = (unsigned short*)alloc((size_t)BS * HDM * 2);
  unsigned short* wbf    = (unsigned short*)alloc((size_t)AHZ * M * SEQ * 2);
  float*          feat   = (float*)         alloc((size_t)M * HDM * 4);
  unsigned short* lnfb   = (unsigned short*)alloc((size_t)M * HDM * 2);
  float*          hbuf   = (float*)         alloc((size_t)M * IFZ * 4);  // ag, then h
  unsigned short* ybf    = (unsigned short*)alloc((size_t)M * IFZ * 2);
  float*          y1     = (float*)         alloc((size_t)M * IFZ * 4);
  unsigned short* y1b    = (unsigned short*)alloc((size_t)M * IFZ * 2);
  float*          y2     = (float*)         alloc((size_t)M * IFZ * 4);
  unsigned short* hbf    = (unsigned short*)alloc((size_t)M * IFZ * 2);
  float*          logits = (float*)         alloc((size_t)M * NRES * 4);
  float*          nf     = (float*)         alloc((size_t)N * IFZ * 4);
  (void)ws_size;

  float* out        = (float*)d_out;
  float* out_logits = out + (size_t)N * IFZ;
  float* out_scores = out + (size_t)N * IFZ + (size_t)N * NRES;

  const int T = 256;
  auto nb = [](long n, int t) { return (unsigned)((n + t - 1) / t); };

  // 0) zero output (logits/scores rows outside prot_mask must be 0) and nf
  zero_f32<<<nb(out_size, T), T, 0, stream>>>(out, (long)out_size);
  zero_f32<<<nb((long)N * IFZ, T), T, 0, stream>>>(nf, (long)N * IFZ);
  make_bidx<<<nb(M, T), T, 0, stream>>>(bat, pm, bidx, M);

  // 1) bf16 staging of weights / activations
  cvt_bf16<<<nb((long)HDM * IFZ, T), T, 0, stream>>>(Wq, Wqb, (long)HDM * IFZ);
  cvt_bf16<<<nb((long)HDM * SFZ, T), T, 0, stream>>>(Wk, Wkb, (long)HDM * SFZ);
  cvt_bf16<<<nb((long)HDM * SFZ, T), T, 0, stream>>>(Wv, Wvb, (long)HDM * SFZ);
  cvt_bf16<<<nb((long)IFZ * HDM, T), T, 0, stream>>>(Wag, Wagb, (long)IFZ * HDM);
  cvt_bf16<<<nb((long)3 * IFZ * IFZ, T), T, 0, stream>>>(rw1, w1b, (long)3 * IFZ * IFZ);
  cvt_bf16<<<nb((long)3 * IFZ * IFZ, T), T, 0, stream>>>(rw2, w2b, (long)3 * IFZ * IFZ);
  cvt_bf16<<<nb((long)NRES * IFZ, T), T, 0, stream>>>(hw, hwb, (long)NRES * IFZ);
  cvt_bf16<<<nb((long)BS * SFZ, T), T, 0, stream>>>(emb, embb, (long)BS * SFZ);
  gather_cvt_x<<<nb((long)M * IFZ, T), T, 0, stream>>>(x, pm, xqb, M);

  // 2) projections (bf16 WMMA, f32 accumulate)
  gemm_bf16<<<dim3((M + 31) / 32, HDM / 16), 32, 0, stream>>>(
      xqb, IFZ, Wqb, IFZ, qf, HDM, M, HDM, IFZ, nullptr, 0, 1.f);
  cvt_bf16<<<nb((long)M * HDM, T), T, 0, stream>>>(qf, qb, (long)M * HDM);
  gemm_bf16<<<dim3((BS + 31) / 32, HDM / 16), 32, 0, stream>>>(
      embb, SFZ, Wkb, SFZ, kf, HDM, BS, HDM, SFZ, nullptr, 0, 1.f);
  cvt_bf16<<<nb((long)BS * HDM, T), T, 0, stream>>>(kf, kb, (long)BS * HDM);
  gemm_bf16<<<dim3((BS + 31) / 32, HDM / 16), 32, 0, stream>>>(
      embb, SFZ, Wvb, SFZ, vf, HDM, BS, HDM, SFZ, nullptr, 0, 1.f);
  v_transpose<<<nb((long)BS * HDM, T), T, 0, stream>>>(vf, vtb, B);

  // 3) attention
  attn_scores<<<dim3((M + 15) / 16, SEQ / 16, AHZ * B), 32, 0, stream>>>(
      qb, kb, bidx, pm, out_scores, M, B);
  softmax_w<<<(unsigned)(M * AHZ), 32, 0, stream>>>(out_scores, pm, bidx, mask, wbf, M);
  attn_feat<<<dim3((M + 15) / 16, AFZ / 16, AHZ * B), 32, 0, stream>>>(
      wbf, vtb, bidx, feat, M, B);

  // 4) aggregate: ag = LN(feat) @ Wag^T
  ln_to_bf16<<<(unsigned)M, 32, 0, stream>>>(feat, HDM, agS, agB, lnfb);
  gemm_bf16<<<dim3((M + 31) / 32, IFZ / 16), 32, 0, stream>>>(
      lnfb, HDM, Wagb, HDM, hbuf, IFZ, M, IFZ, HDM, nullptr, 0, 1.f);
  scatter_rows<<<nb((long)M * IFZ, T), T, 0, stream>>>(hbuf, pm, nf, M, IFZ);

  // 5) residual MLP stack (h lives in hbuf, updated in place)
  for (int i = 0; i < 3; ++i) {
    ln_to_bf16<<<(unsigned)M, 32, 0, stream>>>(hbuf, IFZ, rls + (size_t)i * IFZ,
                                               rlb + (size_t)i * IFZ, ybf);
    gemm_bf16<<<dim3((M + 31) / 32, IFZ / 16), 32, 0, stream>>>(
        ybf, IFZ, w1b + (size_t)i * IFZ * IFZ, IFZ, y1, IFZ, M, IFZ, IFZ,
        rb1 + (size_t)i * IFZ, 1, 1.f);
    cvt_bf16<<<nb((long)M * IFZ, T), T, 0, stream>>>(y1, y1b, (long)M * IFZ);
    gemm_bf16<<<dim3((M + 31) / 32, IFZ / 16), 32, 0, stream>>>(
        y1b, IFZ, w2b + (size_t)i * IFZ * IFZ, IFZ, y2, IFZ, M, IFZ, IFZ,
        rb2 + (size_t)i * IFZ, 0, 1.f);
    resid_add<<<nb((long)M * IFZ, T), T, 0, stream>>>(hbuf, y2, (long)M * IFZ);
  }

  // 6) head logits + scatter
  cvt_bf16<<<nb((long)M * IFZ, T), T, 0, stream>>>(hbuf, hbf, (long)M * IFZ);
  gemm_bf16<<<dim3((M + 31) / 32, (NRES + 15) / 16), 32, 0, stream>>>(
      hbf, IFZ, hwb, IFZ, logits, NRES, M, NRES, IFZ, hb, 0, 1.f);
  scatter_rows<<<nb((long)M * NRES, T), T, 0, stream>>>(logits, pm, out_logits, M, NRES);

  // 7) out_x = LN(x + nf/sqrt(2))
  final_ln<<<(unsigned)N, 32, 0, stream>>>(x, nf, enS, enB, out);
}
